// IBETGLF_27144193311251
// MI455X (gfx1250) — matrix operand order repaired and checked by reading, not compiled, not run on hardware
//
#include <hip/hip_runtime.h>
#include <math.h>

// ---------- types ----------
typedef __attribute__((ext_vector_type(16))) __bf16        v16bf;
typedef __attribute__((ext_vector_type(8)))  float         v8f;
typedef __attribute__((ext_vector_type(8)))  unsigned int  v8u;

static __device__ __forceinline__ __bf16 to_bf16(float f) {
    // round-to-nearest-even f32 -> bf16, pure integer path
    unsigned int u = __builtin_bit_cast(unsigned int, f);
    unsigned int r = (u + 0x7FFFu + ((u >> 16) & 1u)) >> 16;
    unsigned short s = (unsigned short)r;
    return __builtin_bit_cast(__bf16, s);
}

static __device__ __forceinline__ float artanh_clip(float x) {
    x = fminf(fmaxf(x, -1.0f + 1e-5f), 1.0f - 1e-5f);
    return atanhf(x);
}

// ---------- expmap0: H = tanh(||u||) * u / ||u|| ----------
// one block per row, 256 threads (D == 256); emits f32 H, bf16 H, bf16 H^T
__global__ void hgnn_expmap0(const float* __restrict__ U,
                             float* __restrict__ Hf, __bf16* __restrict__ Hb,
                             __bf16* __restrict__ HbT, int D, int N) {
    const int row = blockIdx.x;
    const int t   = threadIdx.x;
    float u = U[(size_t)row * D + t];
    __shared__ float red[256];
    red[t] = u * u;
    __syncthreads();
    for (int s = 128; s > 0; s >>= 1) { if (t < s) red[t] += red[t + s]; __syncthreads(); }
    float n = fmaxf(sqrtf(red[0]), 1e-15f);
    float scale = tanhf(n) / n;
    float h = u * scale;
    Hf[(size_t)row * D + t]  = h;
    __bf16 hb = to_bf16(h);
    Hb[(size_t)row * D + t]  = hb;
    HbT[(size_t)t * N + row] = hb;   // transposed copy for NT use of H
}

// ---------- f32 -> bf16 elementwise ----------
__global__ void hgnn_cvt_bf16(const float* __restrict__ X, __bf16* __restrict__ Y, int n) {
    int i = blockIdx.x * blockDim.x + threadIdx.x;
    if (i < n) Y[i] = to_bf16(X[i]);
}

// ---------- WMMA GEMM (NT): C(MxN) = A(MxK,bf16) * B(NxK,bf16)^T ----------
// One 16x16 output tile per wave; K consumed 32 at a time via v_wmma_f32_16x16x32_bf16.
// All operand fragments load as contiguous dword pairs -> global_load_b128 clauses.
// Optional epilogue copies: Cb = bf16(C) row-major, CbT = bf16(C) transposed (ld = M).
__global__ void hgnn_gemm_nt_wmma(const __bf16* __restrict__ A,
                                  const __bf16* __restrict__ B,
                                  float* __restrict__ C,
                                  __bf16* __restrict__ Cb, __bf16* __restrict__ CbT,
                                  int M, int N, int K) {
    const int lane   = threadIdx.x & 31;
    const int wave   = threadIdx.x >> 5;
    const int tilesN = N >> 4;
    const int waveId = blockIdx.x * (blockDim.x >> 5) + wave;
    const int tileM  = (waveId / tilesN) << 4;
    const int tileN  = (waveId % tilesN) << 4;
    const int hl     = lane >> 4;    // lane half: 0 or 1
    const int mr     = lane & 15;    // row (A) / col (B) within tile

    v8f acc = {};
    for (int k0 = 0; k0 < K; k0 += 32) {
        // A fragment: lane holds row (tileM+mr); 16-bit A layout, K-pairs per VGPR
        v8u au;
#pragma unroll
        for (int vg = 0; vg < 8; ++vg) {
            int kk = k0 + ((vg < 4) ? (vg * 2 + hl * 8) : (16 + (vg - 4) * 2 + hl * 8));
            au[vg] = *(const unsigned int*)(A + (size_t)(tileM + mr) * K + kk);
        }
        // B fragment: lane holds column (tileN+mr); same K order as A (8h+s, 16+8h+s-8)
        v8u bu;
#pragma unroll
        for (int vg = 0; vg < 8; ++vg) {
            int kk = k0 + ((vg < 4) ? (vg * 2 + hl * 8) : (16 + (vg - 4) * 2 + hl * 8));
            bu[vg] = *(const unsigned int*)(B + (size_t)(tileN + mr) * K + kk);
        }
        acc = __builtin_amdgcn_wmma_f32_16x16x32_bf16(
                  false, __builtin_bit_cast(v16bf, au),
                  false, __builtin_bit_cast(v16bf, bu),
                  (short)0, acc, false, false);
    }
    // C/D layout: VGPR r -> row r (+8 for lane half 1), N = lane%16
#pragma unroll
    for (int r = 0; r < 8; ++r) {
        int row = tileM + r + hl * 8;
        int col = tileN + mr;
        float v = acc[r];
        C[(size_t)row * N + col] = v;
        if (Cb)  { Cb[(size_t)row * N + col]  = to_bf16(v); }
        if (CbT) { CbT[(size_t)col * M + row] = to_bf16(v); }
    }
}

// ---------- Mobius rescale: Wh = tanh(|Mx|/|x| * artanh(|x|)) * Mx/|Mx| ----------
// one block per row; also emits ||Wh_row||^2 (needed by pairwise distance)
__global__ void hgnn_mobius(const float* __restrict__ Hf, const float* __restrict__ Mx,
                            __bf16* __restrict__ Whb, float* __restrict__ wn2, int D) {
    const int row = blockIdx.x;
    const int t   = threadIdx.x;
    float x = Hf[(size_t)row * D + t];
    float m = Mx[(size_t)row * D + t];
    __shared__ float red[256];
    red[t] = x * x;
    __syncthreads();
    for (int s = 128; s > 0; s >>= 1) { if (t < s) red[t] += red[t + s]; __syncthreads(); }
    float xn2 = red[0];
    __syncthreads();
    red[t] = m * m;
    __syncthreads();
    for (int s = 128; s > 0; s >>= 1) { if (t < s) red[t] += red[t + s]; __syncthreads(); }
    float mn2 = red[0];

    float xn = fmaxf(sqrtf(xn2), 1e-15f);
    float mn = fmaxf(sqrtf(mn2), 1e-15f);
    float tt = tanhf(mn / xn * artanh_clip(xn));
    float scale = tt / mn;
    Whb[(size_t)row * D + t] = to_bf16(m * scale);
    if (t == 0) wn2[row] = tt * tt;
}

// ---------- pairwise hyperbolic dist^2 + masked softmax, fused per row ----------
// uses Gram matrix G = Wh*Wh^T and row norms; one block (512 thr) per row i
__global__ void hgnn_dist_softmax(const float* __restrict__ G,
                                  const float* __restrict__ wn2,
                                  const unsigned char* __restrict__ Adj,
                                  __bf16* __restrict__ attb, int N) {
    const int i = blockIdx.x;
    const int j = threadIdx.x;
    float xy = -G[(size_t)i * N + j];          // u = -x_i, v = x_j
    float x2 = wn2[i];
    float y2 = wn2[j];
    float a  = 1.0f + 2.0f * xy + y2;
    float b  = 1.0f - x2;
    float num2 = a * a * x2 + 2.0f * a * b * xy + b * b * y2;   // ||a*u + b*v||^2
    float den  = fmaxf(1.0f + 2.0f * xy + x2 * y2, 1e-15f);
    float man  = sqrtf(fmaxf(num2, 0.0f)) / den;
    float d    = 2.0f * artanh_clip(man);
    bool  adj  = Adj[(size_t)i * N + j] != 0;
    float score = adj ? d * d : -10000.0f;

    __shared__ float red[512];
    red[j] = score;
    __syncthreads();
    for (int s = 256; s > 0; s >>= 1) { if (j < s) red[j] = fmaxf(red[j], red[j + s]); __syncthreads(); }
    float rmax = red[0];
    __syncthreads();
    float e = expf(score - rmax);
    red[j] = e;
    __syncthreads();
    for (int s = 256; s > 0; s >>= 1) { if (j < s) red[j] += red[j + s]; __syncthreads(); }
    float rsum = red[0];

    float att = adj ? (e / rsum) : 0.0f;
    attb[(size_t)i * N + j] = to_bf16(att);
}

// ---------- host-side orchestration ----------
extern "C" void kernel_launch(void* const* d_in, const int* in_sizes, int n_in,
                              void* d_out, int out_size, void* d_ws, size_t ws_size,
                              hipStream_t stream) {
    (void)in_sizes; (void)n_in; (void)out_size; (void)ws_size;
    const int N = 512, D = 256, L = 2;

    const float*         H_euc = (const float*)d_in[0];          // 512*256 f32
    const float*         att_W = (const float*)d_in[1];          // 2*256*256 f32
    const unsigned char* Adj   = (const unsigned char*)d_in[2];  // 512*512 bool
    float*               out   = (float*)d_out;                  // 512*256 f32

    // workspace carve-up (all 256B-aligned chunk sizes)
    char* w = (char*)d_ws;
    float*  Hf0  = (float*)w;  w += (size_t)N * D * 4;   // 512 KB
    float*  Hf1  = (float*)w;  w += (size_t)N * D * 4;
    __bf16* Hb0  = (__bf16*)w; w += (size_t)N * D * 2;   // 256 KB
    __bf16* Hb1  = (__bf16*)w; w += (size_t)N * D * 2;
    __bf16* HbT0 = (__bf16*)w; w += (size_t)N * D * 2;   // H^T, D x N
    __bf16* HbT1 = (__bf16*)w; w += (size_t)N * D * 2;
    __bf16* Wb   = (__bf16*)w; w += (size_t)L * D * D * 2;
    float*  Mx   = (float*)w;  w += (size_t)N * D * 4;
    __bf16* Whb  = (__bf16*)w; w += (size_t)N * D * 2;
    float*  wn2  = (float*)w;  w += (size_t)N * 4 + 2048;
    float*  G    = (float*)w;  w += (size_t)N * N * 4;   // 1 MB
    __bf16* atb  = (__bf16*)w; w += (size_t)N * N * 2;

    // H = expmap0(H_euc); also bf16 + bf16-transposed copies for WMMA
    hgnn_expmap0<<<N, D, 0, stream>>>(H_euc, Hf0, Hb0, HbT0, D, N);
    // W (both layers) -> bf16 once
    hgnn_cvt_bf16<<<(L * D * D + 255) / 256, 256, 0, stream>>>(att_W, Wb, L * D * D);

    float* Hf = Hf0; __bf16* Hb = Hb0; __bf16* HbT = HbT0;
    for (int l = 0; l < L; ++l) {
        // Mx = H @ W_l^T  : M=512 N=256 K=256 -> 512 tiles / 8 waves-per-block = 64 blocks
        hgnn_gemm_nt_wmma<<<64, 256, 0, stream>>>(Hb, Wb + (size_t)l * D * D,
                                                  Mx, nullptr, nullptr, N, D, D);
        // Mobius rescale -> Whb (bf16) + ||Wh_i||^2
        hgnn_mobius<<<N, D, 0, stream>>>(Hf, Mx, Whb, wn2, D);
        // Gram G = Wh @ Wh^T : M=N=512 K=256 -> 1024 tiles -> 128 blocks
        hgnn_gemm_nt_wmma<<<128, 256, 0, stream>>>(Whb, Whb, G, nullptr, nullptr, N, N, D);
        // fused distance + mask + softmax + re-mask -> bf16 attention
        hgnn_dist_softmax<<<N, N, 0, stream>>>(G, wn2, Adj, atb, N);
        // H_next = att @ H = att * (H^T)^T : NT GEMM, M=512 "N"=256 K=512 -> 64 blocks
        float*  Cdst = (l == L - 1) ? out : Hf1;
        __bf16* Cb   = (l == L - 1) ? nullptr : Hb1;
        __bf16* CbT  = (l == L - 1) ? nullptr : HbT1;
        hgnn_gemm_nt_wmma<<<64, 256, 0, stream>>>(atb, HbT, Cdst, Cb, CbT, N, D, N);
        Hf = Hf1; Hb = Hb1; HbT = HbT1;
    }
}
